// MambaBlock_50929722196042
// MI455X (gfx1250) — compile-verified, matrix operand order checked by reading, not run on hardware
//
#include <hip/hip_runtime.h>
#include <hip/hip_bf16.h>
#include <math.h>

// Problem constants (match reference)
#define DIM   1024
#define STATE 16
#define INNER 2048          // DIM * EXPAND
#define KSIZE 4
#define BATCH 2
#define SEQ   2048
#define NTOK  (BATCH*SEQ)   // 4096 tokens
#define XRW   (2*INNER)     // 4096 (u | res)

typedef __attribute__((ext_vector_type(16))) __bf16 v16bf;
typedef __attribute__((ext_vector_type(8)))  __bf16 v8bf;
typedef __attribute__((ext_vector_type(8)))  float  v8f;

// Pack two f32 into two bf16 (round-with-bias then truncate) in 3 VALU ops:
// v_add_nc_u32 x2 (rounding bias) + v_perm_b32 (select the two high halves).
__device__ __forceinline__ unsigned int pack2_bf16(float x, float y) {
    union { float f; unsigned int u; } ux, uy;
    ux.f = x; uy.f = y;
    unsigned int rx = ux.u + 0x8000u;
    unsigned int ry = uy.u + 0x8000u;
    // {S0,S1} byte pool: bytes0-3 = S1(rx), bytes4-7 = S0(ry)
    // result: b0=rx[2], b1=rx[3], b2=ry[2], b3=ry[3]  -> {bf16(y), bf16(x)}
    return __builtin_amdgcn_perm(ry, rx, 0x07060302u);
}

// ---------------------------------------------------------------------------
// Tiled WMMA GEMM:  C(MxN) = A(MxK) * W(NxK)^T   (fp32 in memory, bf16 MACs)
// Block = 256 threads (8 waves), tile 128x128, K-step 32.
// Register-double-buffered: next tile's global loads issue before the WMMA
// phase so their latency hides behind 8 WMMAs + 12 ds_load_b128.
// ---------------------------------------------------------------------------
#define BM 128
#define BN 128
#define BKK 32
#define LDT 40   // LDS row stride in bf16 elements (80B, multiple of 16B)

__global__ __launch_bounds__(256) void gemm_wmma_bf16(
    const float* __restrict__ A,   // M x K row-major
    const float* __restrict__ W,   // N x K row-major
    float* __restrict__ C,         // M x N row-major
    int M, int N, int K)
{
    __shared__ unsigned short sA[BM * LDT];
    __shared__ unsigned short sB[BN * LDT];

    const int tid  = threadIdx.x;
    const int lane = tid & 31;
    const int wave = tid >> 5;
    const int tileM = blockIdx.y * BM;
    const int tileN = blockIdx.x * BN;

    const int wr = wave & 3;   // wave row group: 32 rows each
    const int wc = wave >> 2;  // wave col group: 64 cols each

    v8f acc[2][4] = {};

    // staging map: 256 threads load 128x32 f32 per tile (one float4 x 4 rows)
    const int gq = tid & 7;    // float4 column within the 32-wide K slice
    const int gr = tid >> 3;   // base row 0..31 (stride 32)

    const int half = lane >> 4;   // WMMA K-half select
    const int mrow = lane & 15;   // WMMA M/N row within 16

    // ---- prologue: load first K-slice into registers ----
    float4 ra[4], rb[4];
    #pragma unroll
    for (int i = 0; i < 4; ++i) {
        int r = gr + 32 * i;
        ra[i] = *(const float4*)(A + (size_t)(tileM + r) * K + gq * 4);
        rb[i] = *(const float4*)(W + (size_t)(tileN + r) * K + gq * 4);
    }

    for (int k0 = 0; k0 < K; k0 += BKK) {
        // ---- pack current regs -> LDS (bf16) ----
        #pragma unroll
        for (int i = 0; i < 4; ++i) {
            int r = gr + 32 * i;
            *(uint2*)(&sA[r * LDT + gq * 4]) =
                make_uint2(pack2_bf16(ra[i].x, ra[i].y), pack2_bf16(ra[i].z, ra[i].w));
            *(uint2*)(&sB[r * LDT + gq * 4]) =
                make_uint2(pack2_bf16(rb[i].x, rb[i].y), pack2_bf16(rb[i].z, rb[i].w));
        }
        __syncthreads();

        // ---- issue next K-slice loads (wait lands at next iter's pack) ----
        if (k0 + BKK < K) {
            #pragma unroll
            for (int i = 0; i < 4; ++i) {
                int r = gr + 32 * i;
                ra[i] = *(const float4*)(A + (size_t)(tileM + r) * K + (k0 + BKK) + gq * 4);
                rb[i] = *(const float4*)(W + (size_t)(tileN + r) * K + (k0 + BKK) + gq * 4);
            }
        }

        // ---- fragments from LDS (ds_load_b128 x2 each) ----
        // 16-bit A layout (ISA 7.12.2): lane = M row, chunks at K = half*8 and 16+half*8
        v16bf afrag[2];
        #pragma unroll
        for (int i = 0; i < 2; ++i) {
            int row = wr * 32 + i * 16 + mrow;
            union { v16bf v; v8bf h[2]; } ua;
            ua.h[0] = *(const v8bf*)(&sA[row * LDT + half * 8]);
            ua.h[1] = *(const v8bf*)(&sA[row * LDT + 16 + half * 8]);
            afrag[i] = ua.v;
        }
        v16bf bfrag[4];
        #pragma unroll
        for (int j = 0; j < 4; ++j) {
            int row = wc * 64 + j * 16 + mrow;   // N index (row of W)
            union { v16bf v; v8bf h[2]; } ub;
            ub.h[0] = *(const v8bf*)(&sB[row * LDT + half * 8]);
            ub.h[1] = *(const v8bf*)(&sB[row * LDT + 16 + half * 8]);
            bfrag[j] = ub.v;
        }

        // ---- 8 WMMAs: 32x64 output per wave per K-step ----
        #pragma unroll
        for (int i = 0; i < 2; ++i)
            #pragma unroll
            for (int j = 0; j < 4; ++j)
                acc[i][j] = __builtin_amdgcn_wmma_f32_16x16x32_bf16(
                    false, afrag[i], false, bfrag[j], (short)0, acc[i][j], false, false);
        __syncthreads();
    }

    // ---- write back (C/D layout: VGPR r -> M = r + 8*(lane/16), N = lane%16) ----
    const int rowhalf = (lane >> 4) * 8;
    const int ncol    = lane & 15;
    #pragma unroll
    for (int i = 0; i < 2; ++i)
        #pragma unroll
        for (int j = 0; j < 4; ++j) {
            size_t rbase = (size_t)(tileM + wr * 32 + i * 16 + rowhalf);
            int    cidx  = tileN + wc * 64 + j * 16 + ncol;
            #pragma unroll
            for (int r = 0; r < 8; ++r)
                C[(rbase + r) * N + cidx] = acc[i][j][r];
        }
}

// ---------------------------------------------------------------------------
// Depthwise causal conv (K=4) + bias.  u[tok,c] from xr[:, :INNER]
// ---------------------------------------------------------------------------
__global__ __launch_bounds__(256) void conv_kernel(
    const float* __restrict__ xr, const float* __restrict__ cw,
    const float* __restrict__ cb, float* __restrict__ u)
{
    int idx = blockIdx.x * blockDim.x + threadIdx.x;   // NTOK*INNER
    int c   = idx % INNER;
    int tok = idx / INNER;
    int b = tok / SEQ, t = tok % SEQ;
    float acc = cb[c];
    #pragma unroll
    for (int k = 0; k < KSIZE; ++k) {
        int tt = t - (KSIZE - 1) + k;
        if (tt >= 0)
            acc = fmaf(cw[c * KSIZE + k], xr[((size_t)b * SEQ + tt) * XRW + c], acc);
    }
    u[idx] = acc;
}

// ---------------------------------------------------------------------------
// ssm = u @ W_x^T   (N = 2*STATE+1 = 33, too skinny for WMMA)
// ---------------------------------------------------------------------------
__global__ __launch_bounds__(64) void ssm_kernel(
    const float* __restrict__ u, const float* __restrict__ Wx,
    float* __restrict__ ssm)
{
    int tok = blockIdx.x;
    int j   = threadIdx.x;
    if (j >= 2 * STATE + 1) return;
    const float* ur = u  + (size_t)tok * INNER;
    const float* wr = Wx + (size_t)j * INNER;
    float acc = 0.f;
    for (int k = 0; k < INNER; k += 4) {
        float4 a = *(const float4*)(ur + k);
        float4 w = *(const float4*)(wr + k);
        acc = fmaf(a.x, w.x, acc); acc = fmaf(a.y, w.y, acc);
        acc = fmaf(a.z, w.z, acc); acc = fmaf(a.w, w.w, acc);
    }
    ssm[(size_t)tok * 33 + j] = acc;
}

// ---------------------------------------------------------------------------
// Sequential SSM scan: one thread per (batch, channel).
// delta computed inline: softplus(ssm[0:16] . W_dt[c] + b_dt[c]).
// h[16] in registers; fused silu(res) gating.
// ---------------------------------------------------------------------------
__global__ __launch_bounds__(256) void scan_kernel(
    const float* __restrict__ u,    const float* __restrict__ ssm,
    const float* __restrict__ xr,   const float* __restrict__ Wdt,
    const float* __restrict__ bdt,  const float* __restrict__ A_log,
    float* __restrict__ ymod)
{
    int gid = blockIdx.x * blockDim.x + threadIdx.x;  // BATCH*INNER
    int c = gid % INNER;
    int b = gid / INNER;

    float Arow[STATE], h[STATE], wdt[STATE];
    #pragma unroll
    for (int s = 0; s < STATE; ++s) {
        Arow[s] = -__expf(A_log[c * STATE + s]);
        wdt[s]  = Wdt[c * STATE + s];
        h[s]    = 0.f;
    }
    const float bias = bdt[c];

    for (int t = 0; t < SEQ; ++t) {
        size_t tok = (size_t)b * SEQ + t;
        const float* sp = ssm + tok * 33;

        // delta = softplus(dt_in . W_dt[c] + b_dt[c])
        float dpre = bias;
        #pragma unroll
        for (int s = 0; s < STATE; ++s) dpre = fmaf(sp[s], wdt[s], dpre);
        float d = (dpre > 20.f) ? dpre : __logf(1.f + __expf(dpre));

        float uu = u[tok * INNER + c];
        float y  = 0.f;
        #pragma unroll
        for (int s = 0; s < STATE; ++s) {
            h[s] = fmaf(h[s], __expf(d * Arow[s]), uu * sp[STATE + s]);
            y += h[s];
        }
        y *= sp[2 * STATE];                       // * C_t

        float r   = xr[tok * XRW + INNER + c];    // res gate
        float sil = r / (1.f + __expf(-r));
        ymod[tok * INNER + c] = y * sil;
    }
}

// ---------------------------------------------------------------------------
extern "C" void kernel_launch(void* const* d_in, const int* in_sizes, int n_in,
                              void* d_out, int out_size, void* d_ws, size_t ws_size,
                              hipStream_t stream) {
    const float* x      = (const float*)d_in[0];
    const float* W_in   = (const float*)d_in[1];
    const float* conv_w = (const float*)d_in[2];
    const float* conv_b = (const float*)d_in[3];
    const float* W_x    = (const float*)d_in[4];
    const float* W_dt   = (const float*)d_in[5];
    const float* b_dt   = (const float*)d_in[6];
    const float* W_out  = (const float*)d_in[7];
    const float* A_log  = (const float*)d_in[8];
    float* out = (float*)d_out;

    float* ws   = (float*)d_ws;
    float* xr   = ws;                                  // NTOK*XRW   (64 MB)
    float* u    = xr   + (size_t)NTOK * XRW;           // NTOK*INNER (32 MB)
    float* ssm  = u    + (size_t)NTOK * INNER;         // NTOK*33
    float* ymod = ssm  + (size_t)NTOK * 33;            // NTOK*INNER (32 MB)

    dim3 blk(256);

    // 1) xr = x @ W_in^T            (M=4096, N=4096, K=1024)  -- WMMA bf16
    gemm_wmma_bf16<<<dim3(XRW / BN, NTOK / BM), blk, 0, stream>>>(
        x, W_in, xr, NTOK, XRW, DIM);

    // 2) depthwise causal conv + bias
    conv_kernel<<<(NTOK * INNER) / 256, blk, 0, stream>>>(xr, conv_w, conv_b, u);

    // 3) ssm = u @ W_x^T            (N=33, skinny -> VALU dots)
    ssm_kernel<<<NTOK, 64, 0, stream>>>(u, W_x, ssm);

    // 4) recurrent scan + inline softplus(delta) + silu(res) gate
    scan_kernel<<<(BATCH * INNER) / 256, blk, 0, stream>>>(
        u, ssm, xr, W_dt, b_dt, A_log, ymod);

    // 5) out = ymod @ W_out^T       (M=4096, N=1024, K=2048)  -- WMMA bf16
    gemm_wmma_bf16<<<dim3(DIM / BN, NTOK / BM), blk, 0, stream>>>(
        ymod, W_out, out, NTOK, DIM, INNER);
}
